// Decoder_61237643706389
// MI455X (gfx1250) — matrix operand order repaired and checked by reading, not compiled, not run on hardware
//
#include <hip/hip_runtime.h>
#include <math.h>

typedef __attribute__((ext_vector_type(16))) _Float16 v16h;
typedef __attribute__((ext_vector_type(8)))  _Float16 v8h;
typedef __attribute__((ext_vector_type(8)))  float    v8f;

#define Gc 32
#define Cc 96
#define NPT 32768           // 32^3 query points per batch
#define OUTC 45

// packed-weight offsets (in halfs)
#define OFF_W00 0
#define OFF_W01 18432
#define OFF_W02 27648
#define OFF_W10 36864
#define OFF_W11 55296
#define OFF_W12 64512
#define OFF_WPOST 73728
#define WPK_HALFS 78336
#define WPK_BYTES (WPK_HALFS*2)     // 156672
#define INV_OFF  WPK_BYTES
#define CVT_OFF  (WPK_BYTES + 256)  // f16 transposed context in ws

// dynamic LDS layout
#define WAVES 8
#define PW_BYTES 6912               // per-wave scratch: actH 6144 + misc 768
#define DYN_BYTES (WPK_BYTES + WAVES*PW_BYTES)   // 211968

// ---------------------------------------------------------------- pre-kernels

__global__ void invert4x4_kernel(const float* __restrict__ A, float* __restrict__ Ainv, int nb) {
    int b = blockIdx.x * blockDim.x + threadIdx.x;
    if (b >= nb) return;
    float M[4][8];
    for (int r = 0; r < 4; ++r)
        for (int c = 0; c < 4; ++c) {
            M[r][c]     = A[b*16 + r*4 + c];
            M[r][4 + c] = (r == c) ? 1.f : 0.f;
        }
    for (int col = 0; col < 4; ++col) {
        int piv = col; float best = fabsf(M[col][col]);
        for (int r = col + 1; r < 4; ++r) {
            float v = fabsf(M[r][col]);
            if (v > best) { best = v; piv = r; }
        }
        if (piv != col)
            for (int c = 0; c < 8; ++c) { float t = M[col][c]; M[col][c] = M[piv][c]; M[piv][c] = t; }
        float inv = 1.f / M[col][col];
        for (int c = 0; c < 8; ++c) M[col][c] *= inv;
        for (int r = 0; r < 4; ++r) {
            if (r == col) continue;
            float f = M[r][col];
            for (int c = 0; c < 8; ++c) M[r][c] -= f * M[col][c];
        }
    }
    for (int r = 0; r < 4; ++r)
        for (int c = 0; c < 4; ++c)
            Ainv[b*16 + r*4 + c] = M[r][4 + c];
}

// context_v [B,C,32,32,32] f32 -> cvt [B,32^3,C] f16
__global__ void transpose_ctx_kernel(const float* __restrict__ cv, _Float16* __restrict__ cvt, int total) {
    int i = blockIdx.x * 256 + threadIdx.x;
    if (i >= total) return;
    int c    = i % Cc;
    int rest = i / Cc;
    int vox  = rest & (NPT - 1);
    int b    = rest >> 15;
    cvt[i] = (_Float16)cv[(((size_t)b * Cc + c) << 15) + vox];
}

// pack weights into WMMA B-fragment order (f16).
// fragment element t (0..511): lane = t>>4, h = t&15
//   k = kc*32 + (lane>=16 ? 8 : 0) + (h&7) + (h>=8 ? 16 : 0) ;  n = nt*16 + (lane&15)
__global__ void pack_weights_kernel(const float* __restrict__ w00, const float* __restrict__ w01,
                                    const float* __restrict__ w02, const float* __restrict__ w10,
                                    const float* __restrict__ w11, const float* __restrict__ w12,
                                    const float* __restrict__ wpost, _Float16* __restrict__ dst) {
    const int Ks[7]  = {174, 96, 96, 174, 96, 96, 96};
    const int Ns[7]  = {96, 96, 96, 96, 96, 96, 45};
    const int KC[7]  = {6, 3, 3, 6, 3, 3, 3};
    const int NT[7]  = {6, 6, 6, 6, 6, 6, 3};
    const int OFF[7] = {OFF_W00, OFF_W01, OFF_W02, OFF_W10, OFF_W11, OFF_W12, OFF_WPOST};
    int layer = blockIdx.y;
    int frag  = blockIdx.x;
    if (frag >= KC[layer] * NT[layer]) return;
    const float* W = (layer == 0) ? w00 : (layer == 1) ? w01 : (layer == 2) ? w02 :
                     (layer == 3) ? w10 : (layer == 4) ? w11 : (layer == 5) ? w12 : wpost;
    int t    = threadIdx.x;
    int kc   = frag / NT[layer];
    int nt   = frag % NT[layer];
    int lane = t >> 4;
    int h    = t & 15;
    int k    = kc * 32 + ((lane >> 4) << 3) + (h & 7) + ((h >> 3) << 4);
    int n    = nt * 16 + (lane & 15);
    float v  = (k < Ks[layer] && n < Ns[layer]) ? W[k * Ns[layer] + n] : 0.f;
    dst[OFF[layer] + frag * 512 + t] = (_Float16)v;
}

// ---------------------------------------------------------------- main kernel

// A operand (16x32 f16): lane half 'half' holds k_local = half*8 + {0..7, 16..23}
// -> two contiguous 8-half LDS loads per chunk from row buffer (stride 192 halfs).
__device__ __forceinline__ v16h a_from_row(const _Float16* __restrict__ rowbuf, int p, int half, int kc) {
    const _Float16* base = rowbuf + p * 192 + kc * 32 + half * 8;
    v8h lo = *(const v8h*)(base);
    v8h hi = *(const v8h*)(base + 16);
    v16h a;
#pragma unroll
    for (int h = 0; h < 8; ++h) { a[h] = lo[h]; a[h + 8] = hi[h]; }
    return a;
}

template <int NT>
__device__ __forceinline__ void gemm_tiles(v8f (&c)[NT], v16h a, const _Float16* __restrict__ wf, int lane) {
#pragma unroll
    for (int nt = 0; nt < NT; ++nt) {
        v16h bf = *(const v16h*)(wf + nt * 512 + lane * 16);   // 32B contiguous, LDS
        c[nt] = __builtin_amdgcn_wmma_f32_16x16x32_f16(false, a, false, bf, (short)0, c[nt], false, false);
    }
}

template <int NT>
__device__ __forceinline__ void bias_silu(v8f (&h)[NT], const float* __restrict__ bias, int ln) {
#pragma unroll
    for (int nt = 0; nt < NT; ++nt) {
        float bb = bias[nt * 16 + ln];
#pragma unroll
        for (int r = 0; r < 8; ++r) {
            float x = h[nt][r] + bb;
            h[nt][r] = x * (1.0f / (1.0f + __expf(-x)));
        }
    }
}

// stage C-layout f32 regs into f16 row buffer cols 0..95
__device__ __forceinline__ void stage_act(_Float16* __restrict__ rowbuf, const v8f (&y)[6], int ln, int half) {
#pragma unroll
    for (int nt = 0; nt < 6; ++nt)
#pragma unroll
        for (int r = 0; r < 8; ++r)
            rowbuf[(r + 8 * half) * 192 + nt * 16 + ln] = (_Float16)y[nt][r];
}

__global__ __launch_bounds__(256) void decoder_kernel(
    const float* __restrict__ query, const unsigned char* __restrict__ qmask,
    const float* __restrict__ grid3,
    const float* __restrict__ b00, const float* __restrict__ b01, const float* __restrict__ b02,
    const float* __restrict__ b10, const float* __restrict__ b11, const float* __restrict__ b12,
    const float* __restrict__ bpost,
    const float* __restrict__ invA, const _Float16* __restrict__ wpk,
    const _Float16* __restrict__ cvt, float* __restrict__ out) {

    extern __shared__ char dyn[];
    _Float16* ldsW = (_Float16*)dyn;                 // 156672 B packed weights

    const int tid  = threadIdx.x;
    const int lane = tid & 31;
    const int wv   = tid >> 5;
    const int half = lane >> 4;
    const int ln   = lane & 15;

    char* wb = dyn + WPK_BYTES + wv * PW_BYTES;
    _Float16* actH  = (_Float16*)wb;                 // [16][192] f16: cols 0..95 act, 96..191 coord
    float*    sQW   = (float*)(wb + 6144);           // [16][3]
    float*    sSub  = (float*)(wb + 6336);           // [16][3]
    int*      sBot  = (int*)(wb + 6528);             // [16][3]
    float*    sMask = (float*)(wb + 6720);           // [16]
    int*      sVox  = (int*)(wb + 6784);             // [16]
    float*    sWtri = (float*)(wb + 6848);           // [16]

    // ---- stage packed weights global -> LDS (whole block, once)
    {
        const unsigned* gw = (const unsigned*)wpk;
        unsigned* lw = (unsigned*)ldsW;
        for (int i = tid; i < WPK_BYTES / 4; i += 256)
            lw[i] = gw[i];
    }

    const int waveGlobal = blockIdx.x * WAVES + wv;
    const int b      = waveGlobal >> 11;             // 2048 waves per batch
    const int ptBase = (waveGlobal & 2047) << 4;

    // ---- per-point setup (corner independent)
    if (lane < 16) {
        int p  = lane;
        int gp = ptBase + p;
        const float* qp = query + ((size_t)b * NPT + gp) * 3;
        const float* qd = query + ((size_t)b * NPT + (NPT / 2)) * 3;
        float m  = qmask[(size_t)b * NPT + gp] ? 1.f : 0.f;
        float qx = (m != 0.f) ? qp[0] : qd[0];
        float qy = (m != 0.f) ? qp[1] : qd[1];
        float qz = (m != 0.f) ? qp[2] : qd[2];
        const float* A = invA + b * 16;
        float vx = A[0] * qx + A[1] * qy + A[2]  * qz + A[3];
        float vy = A[4] * qx + A[5] * qy + A[6]  * qz + A[7];
        float vz = A[8] * qx + A[9] * qy + A[10] * qz + A[11];
        float fx = floorf(vx), fy = floorf(vy), fz = floorf(vz);
        sQW[p*3+0] = qx; sQW[p*3+1] = qy; sQW[p*3+2] = qz;
        sSub[p*3+0] = vx - fx; sSub[p*3+1] = vy - fy; sSub[p*3+2] = vz - fz;
        sBot[p*3+0] = (int)fx; sBot[p*3+1] = (int)fy; sBot[p*3+2] = (int)fz;
        sMask[p] = m;
    }
    __syncthreads();

    v8f accP[3];
#pragma unroll
    for (int nt = 0; nt < 3; ++nt)
#pragma unroll
        for (int r = 0; r < 8; ++r) accP[nt][r] = 0.f;

    for (int co = 0; co < 8; ++co) {
        const int oi = (co >> 2) & 1, oj = (co >> 1) & 1, ok = co & 1;

        if (lane < 16) {
            int p  = lane;
            int ix = sBot[p*3+0] + oi;
            int iy = sBot[p*3+1] + oj;
            int iz = sBot[p*3+2] + ok;
            sVox[p] = (ix * Gc + iy) * Gc + iz;
            sWtri[p] = fabsf((1.f - oi) - sSub[p*3+0]) *
                       fabsf((1.f - oj) - sSub[p*3+1]) *
                       fabsf((1.f - ok) - sSub[p*3+2]);
        }
        __syncthreads();

        // ---- coord vector into actH cols 96..191 : [cw(3), qw(3), enc(72), pad]
        {
            int p = ln;
            float m = sMask[p];
            int e0 = half * 48;
            for (int t = 0; t < 48; ++t) {
                int e = e0 + t;
                float val;
                if (e < 3) {
                    val = grid3[((size_t)b * NPT + sVox[p]) * 3 + e];
                } else if (e < 6) {
                    val = sQW[p*3 + e - 3];
                } else if (e < 78) {
                    int ee = e - 6, d = ee / 24, rr = ee % 24, j = rr % 12;
                    float off = (d == 0) ? (float)oi : (d == 1) ? (float)oj : (float)ok;
                    float rel = (sSub[p*3+d] - off + 1.f) * 0.5f;
                    float freq = __expf((float)j * 0.09155102862f); // ln(3)/12
                    float ang  = 6.283185307f * rel * freq;
                    val = (rr < 12) ? __sinf(ang) : __cosf(ang);
                } else {
                    val = 0.f;
                }
                actH[p * 192 + 96 + e] = (_Float16)(val * m);
            }
        }
        __syncthreads();

        // ---- features -> y in C-register layout (f32)
        int   voxm[8];
        float mm[8], wt[8];
#pragma unroll
        for (int r = 0; r < 8; ++r) {
            int m8 = r + 8 * half;
            voxm[r] = sVox[m8];
            mm[r]   = sMask[m8];
            wt[r]   = sWtri[m8];
        }
        v8f yv[6];
#pragma unroll
        for (int nt = 0; nt < 6; ++nt)
#pragma unroll
            for (int r = 0; r < 8; ++r)
                yv[nt][r] = (float)cvt[((size_t)b * NPT + voxm[r]) * Cc + nt * 16 + ln] * mm[r];

        // ---- two SkipMLP blocks
#pragma unroll
        for (int blk = 0; blk < 2; ++blk) {
            const _Float16* wA = ldsW + (blk ? OFF_W10 : OFF_W00);
            const _Float16* wB = ldsW + (blk ? OFF_W11 : OFF_W01);
            const _Float16* wC = ldsW + (blk ? OFF_W12 : OFF_W02);
            const float* bA = blk ? b10 : b00;
            const float* bB = blk ? b11 : b01;
            const float* bC = blk ? b12 : b02;

            stage_act(actH, yv, ln, half);
            __syncthreads();

            v8f h1[6];
#pragma unroll
            for (int nt = 0; nt < 6; ++nt)
#pragma unroll
                for (int r = 0; r < 8; ++r) h1[nt][r] = 0.f;
#pragma unroll
            for (int kc = 0; kc < 6; ++kc) {   // kc 0..2: act, kc 3..5: coord (same buffer)
                v16h a = a_from_row(actH, ln, half, kc);
                gemm_tiles<6>(h1, a, wA + kc * 6 * 512, lane);
            }
            bias_silu<6>(h1, bA, ln);
            __syncthreads();
            stage_act(actH, h1, ln, half);
            __syncthreads();

            v8f h2[6];
#pragma unroll
            for (int nt = 0; nt < 6; ++nt)
#pragma unroll
                for (int r = 0; r < 8; ++r) h2[nt][r] = 0.f;
#pragma unroll
            for (int kc = 0; kc < 3; ++kc) {
                v16h a = a_from_row(actH, ln, half, kc);
                gemm_tiles<6>(h2, a, wB + kc * 6 * 512, lane);
            }
            bias_silu<6>(h2, bB, ln);
            __syncthreads();
            stage_act(actH, h2, ln, half);
            __syncthreads();

            v8f h3[6];
#pragma unroll
            for (int nt = 0; nt < 6; ++nt)
#pragma unroll
                for (int r = 0; r < 8; ++r) h3[nt][r] = 0.f;
#pragma unroll
            for (int kc = 0; kc < 3; ++kc) {
                v16h a = a_from_row(actH, ln, half, kc);
                gemm_tiles<6>(h3, a, wC + kc * 6 * 512, lane);
            }
            bias_silu<6>(h3, bC, ln);
#pragma unroll
            for (int nt = 0; nt < 6; ++nt)
#pragma unroll
                for (int r = 0; r < 8; ++r) yv[nt][r] += h3[nt][r];
            __syncthreads();
        }

        // ---- post head (96 -> 48 padded) + trilinear accumulate
        stage_act(actH, yv, ln, half);
        __syncthreads();
        v8f pr[3];
#pragma unroll
        for (int nt = 0; nt < 3; ++nt)
#pragma unroll
            for (int r = 0; r < 8; ++r) pr[nt][r] = 0.f;
#pragma unroll
        for (int kc = 0; kc < 3; ++kc) {
            v16h a = a_from_row(actH, ln, half, kc);
            gemm_tiles<3>(pr, a, ldsW + OFF_WPOST + kc * 3 * 512, lane);
        }
#pragma unroll
        for (int nt = 0; nt < 3; ++nt) {
            int n = nt * 16 + ln;
            float bb = (n < OUTC) ? bpost[n] : 0.f;
#pragma unroll
            for (int r = 0; r < 8; ++r)
                accP[nt][r] += wt[r] * (pr[nt][r] + bb);
        }
        __syncthreads();
    }

    // ---- write out: [B, 45, 32^3]
#pragma unroll
    for (int nt = 0; nt < 3; ++nt) {
        int o = nt * 16 + ln;
        if (o < OUTC) {
#pragma unroll
            for (int r = 0; r < 8; ++r) {
                int m8 = r + 8 * half;
                out[(((size_t)b * OUTC + o) << 15) + ptBase + m8] = accP[nt][r];
            }
        }
    }
}

// ---------------------------------------------------------------- launch

extern "C" void kernel_launch(void* const* d_in, const int* in_sizes, int n_in,
                              void* d_out, int out_size, void* d_ws, size_t ws_size,
                              hipStream_t stream) {
    const float* context_v = (const float*)d_in[0];
    const float* grid3     = (const float*)d_in[1];
    const float* query     = (const float*)d_in[2];
    const unsigned char* qmask = (const unsigned char*)d_in[3];
    const float* affine    = (const float*)d_in[4];
    const float* w00 = (const float*)d_in[8];
    const float* b00 = (const float*)d_in[9];
    const float* w01 = (const float*)d_in[10];
    const float* b01 = (const float*)d_in[11];
    const float* w02 = (const float*)d_in[12];
    const float* b02 = (const float*)d_in[13];
    const float* w10 = (const float*)d_in[14];
    const float* b10 = (const float*)d_in[15];
    const float* w11 = (const float*)d_in[16];
    const float* b11 = (const float*)d_in[17];
    const float* w12 = (const float*)d_in[18];
    const float* b12 = (const float*)d_in[19];
    const float* wpost = (const float*)d_in[20];
    const float* bpost = (const float*)d_in[21];

    const int B = in_sizes[0] / (Cc * NPT);

    char* ws = (char*)d_ws;
    _Float16* wpk = (_Float16*)ws;
    float*    invA = (float*)(ws + INV_OFF);
    _Float16* cvt  = (_Float16*)(ws + CVT_OFF);

    invert4x4_kernel<<<1, 32, 0, stream>>>(affine, invA, B);

    const int totalCvt = B * NPT * Cc;
    transpose_ctx_kernel<<<(totalCvt + 255) / 256, 256, 0, stream>>>(context_v, cvt, totalCvt);

    pack_weights_kernel<<<dim3(36, 7), 512, 0, stream>>>(w00, w01, w02, w10, w11, w12, wpost, wpk);

    const int totalWaves = B * (NPT / 16);
    const int blocks = totalWaves / WAVES;
    decoder_kernel<<<blocks, 256, DYN_BYTES, stream>>>(query, qmask, grid3,
                                                       b00, b01, b02, b10, b11, b12, bpost,
                                                       invA, wpk, cvt, (float*)d_out);
}
